// GNN_83013127897195
// MI455X (gfx1250) — matrix operand order repaired and checked by reading, not compiled, not run on hardware
//
#include <hip/hip_runtime.h>

#define HIDD 200
#define NRELD 16
#define NBLK 5
#define BSZ 40

typedef __attribute__((ext_vector_type(16))) _Float16 v16h;
typedef __attribute__((ext_vector_type(8)))  _Float16 v8h;
typedef __attribute__((ext_vector_type(8)))  float    v8f;

// ---------------------------------------------------------------------------
// WMMA GEMM, register-blocked over N: one wave computes a 16 x (NT*16) strip.
//   C[m, c_off+n] (+)= sum_k A[m, a_off+k] * BT[n, k]      (BT = B^T)
// A: f16 row-major (lda); BT: f16 row-major (ldbt), zero-padded to Kpad cols
// and >= NT*16 rows; C: f32 row-major (ldc). The A fragment is loaded once
// per K-step and reused by NT WMMAs (distinct D accumulators -> no RAW NOPs).
// blockIdx.z advances per-matrix offsets (the 5 diagonal blocks).
// ---------------------------------------------------------------------------
template <int NT>
__global__ __launch_bounds__(32)
void rgcn_wmma_gemm_f16(const _Float16* __restrict__ A, int lda, int a_off,
                        const _Float16* __restrict__ BT, int ldbt,
                        float* __restrict__ C, int ldc, int c_off,
                        int Nc, int Kpad, int accumulate,
                        int zA, int zBT, int zC)
{
    const int z    = blockIdx.z;
    a_off += z * zA;
    c_off += z * zC;
    BT    += (size_t)z * (size_t)zBT;

    const int lane = threadIdx.x & 31;
    const int g    = lane >> 4;     // half-wave group
    const int lr   = lane & 15;     // row (A) / col (B,C) within tile
    const int m0   = blockIdx.x * 16;
    const int n0   = blockIdx.y * (NT * 16);

    v8f acc[NT];
    if (accumulate) {
#pragma unroll
        for (int t = 0; t < NT; ++t)
#pragma unroll
            for (int i = 0; i < 8; ++i) {
                const int row = m0 + i + 8 * g;
                const int col = n0 + t * 16 + lr;
                acc[t][i] = (col < Nc) ? C[(size_t)row * ldc + c_off + col] : 0.0f;
            }
    } else {
#pragma unroll
        for (int t = 0; t < NT; ++t)
#pragma unroll
            for (int i = 0; i < 8; ++i) acc[t][i] = 0.0f;
    }

    const _Float16* arow  = A + (size_t)(m0 + lr) * lda + a_off;
    const _Float16* brow0 = BT + (size_t)(n0 + lr) * ldbt;

    for (int k0 = 0; k0 < Kpad; k0 += 32) {
        // A fragment: lanes 0-15 row m, K = k0+[0..7] & k0+16+[0..7]; lanes 16-31: +8
        v8h alo = *(const v8h*)(arow + k0 + 8 * g);
        v8h ahi = *(const v8h*)(arow + k0 + 16 + 8 * g);
        v16h a;
#pragma unroll
        for (int i = 0; i < 8; ++i) { a[i] = alo[i]; a[i + 8] = ahi[i]; }
#pragma unroll
        for (int t = 0; t < NT; ++t) {
            // B fragment: column n of B == row n of BT, K = k0+8*g+[0..7] & k0+16+8*g+[0..7]
            const _Float16* bp = brow0 + (size_t)(t * 16) * ldbt + k0;
            v8h blo = *(const v8h*)(bp + 8 * g);
            v8h bhi = *(const v8h*)(bp + 16 + 8 * g);
            v16h b;
#pragma unroll
            for (int i = 0; i < 8; ++i) { b[i] = blo[i]; b[i + 8] = bhi[i]; }
            acc[t] = __builtin_amdgcn_wmma_f32_16x16x32_f16(
                         false, a, false, b, (short)0, acc[t], false, false);
        }
    }

#pragma unroll
    for (int t = 0; t < NT; ++t)
#pragma unroll
        for (int i = 0; i < 8; ++i) {
            const int row = m0 + i + 8 * g;
            const int col = n0 + t * 16 + lr;
            if (col < Nc) C[(size_t)row * ldc + c_off + col] = acc[t][i];
        }
}

// ---------------------------------------------------------------------------
// Elementwise / helper kernels
// ---------------------------------------------------------------------------
__global__ void rgcn_fill_f32(float* p, float v, long n) {
    long i = (long)blockIdx.x * blockDim.x + threadIdx.x;
    if (i < n) p[i] = v;
}

__global__ void rgcn_fill_f16(_Float16* p, long n) {
    long i = (long)blockIdx.x * blockDim.x + threadIdx.x;
    if (i < n) p[i] = (_Float16)0.0f;
}

__global__ void rgcn_cvt_f16(const float* __restrict__ s, _Float16* __restrict__ d, long n) {
    long i = (long)blockIdx.x * blockDim.x + threadIdx.x;
    if (i < n) d[i] = (_Float16)s[i];
}

__global__ void rgcn_relu_cvt(const float* __restrict__ s, _Float16* __restrict__ d, long n) {
    long i = (long)blockIdx.x * blockDim.x + threadIdx.x;
    if (i < n) { float v = s[i]; d[i] = (_Float16)(v > 0.0f ? v : 0.0f); }
}

// dst[z][n][k] = (f16) src[z][k][n]   (K x Nc row-major source, padded dest ld)
__global__ void rgcn_transpose_cvt(const float* __restrict__ src, _Float16* __restrict__ dst,
                                   int K, int Nc, int ld, long srcStride, long dstStride) {
    long z = blockIdx.z;
    src += z * srcStride;
    dst += z * dstStride;
    int i = blockIdx.x * blockDim.x + threadIdx.x;
    if (i < K * Nc) {
        int k = i / Nc, n = i % Nc;
        dst[(size_t)n * ld + k] = (_Float16)src[i];
    }
}

__global__ void rgcn_init_bias(float* __restrict__ out, const float* __restrict__ bias, long n) {
    long i = (long)blockIdx.x * blockDim.x + threadIdx.x;
    if (i < n) out[i] = bias[(int)(i % HIDD)];
}

__global__ void rgcn_deg_count(const int* __restrict__ dstv, const int* __restrict__ et,
                               float* __restrict__ deg, int E) {
    int e = blockIdx.x * blockDim.x + threadIdx.x;
    if (e < E) atomicAdd(&deg[(size_t)dstv[e] * NRELD + et[e]], 1.0f);
}

// One wave per edge: out[dst] += Hr[src] / deg(dst, rel)  for edges of `rel`
__global__ __launch_bounds__(256)
void rgcn_scatter_rel(const int* __restrict__ srcv, const int* __restrict__ dstv,
                      const int* __restrict__ et, const float* __restrict__ deg,
                      const float* __restrict__ Hr, float* __restrict__ out,
                      int rel, int E)
{
    int e    = blockIdx.x * (blockDim.x >> 5) + (threadIdx.x >> 5);
    int lane = threadIdx.x & 31;
    if (e >= E) return;
    if (et[e] != rel) return;
    int s = srcv[e], d = dstv[e];
    float inv = 1.0f / deg[(size_t)d * NRELD + rel];
    const float* hr = Hr + (size_t)s * HIDD;
    float* o = out + (size_t)d * HIDD;
    for (int h = lane; h < HIDD; h += 32)
        atomicAdd(&o[h], hr[h] * inv);
}

// One wave per edge: scores[e] = sum_h z[src][h]*rel_emb[et][h]*z[dst][h]
__global__ __launch_bounds__(256)
void rgcn_decode(const int* __restrict__ srcv, const int* __restrict__ dstv,
                 const int* __restrict__ et, const float* __restrict__ z,
                 const float* __restrict__ rel_emb, float* __restrict__ scores, int E)
{
    int e    = blockIdx.x * (blockDim.x >> 5) + (threadIdx.x >> 5);
    int lane = threadIdx.x & 31;
    if (e >= E) return;
    int s = srcv[e], d = dstv[e], r = et[e];
    const float* zs = z + (size_t)s * HIDD;
    const float* zd = z + (size_t)d * HIDD;
    const float* re = rel_emb + (size_t)r * HIDD;
    float acc = 0.0f;
    for (int h = lane; h < HIDD; h += 32) acc += zs[h] * re[h] * zd[h];
#pragma unroll
    for (int off = 16; off > 0; off >>= 1) acc += __shfl_xor(acc, off, 32);
    if (lane == 0) scores[e] = acc;
}

// ---------------------------------------------------------------------------
static inline long cdivl(long a, long b) { return (a + b - 1) / b; }

extern "C" void kernel_launch(void* const* d_in, const int* in_sizes, int n_in,
                              void* d_out, int out_size, void* d_ws, size_t ws_size,
                              hipStream_t stream)
{
    const int*   edge_index = (const int*)d_in[0];   // [2, E]
    const int*   edge_type  = (const int*)d_in[1];   // [E]
    const float* node_emb   = (const float*)d_in[2]; // [N, 200]
    const float* W1         = (const float*)d_in[3]; // [16,5,40,40]
    const float* root1      = (const float*)d_in[4]; // [200,200]
    const float* b1         = (const float*)d_in[5];
    const float* W2         = (const float*)d_in[6];
    const float* root2      = (const float*)d_in[7];
    const float* b2         = (const float*)d_in[8];
    const float* rel_emb    = (const float*)d_in[9]; // [16,200]

    const int E = in_sizes[1];
    const int N = in_sizes[2] / HIDD;
    const int* srcv = edge_index;
    const int* dstv = edge_index + E;

    // --- workspace carve-out (256B-aligned; slack covers K-pad overruns) ---
    char* p = (char*)d_ws;
    size_t used = 0;
    auto take = [&](size_t bytes) -> void* {
        void* r = p + used;
        used += (bytes + 255) & ~(size_t)255;
        return r;
    };
    _Float16* x_h  = (_Float16*)take((size_t)N * HIDD * 2 + 256); // layer input, f16
    float*    Hr   = (float*)   take((size_t)N * HIDD * 4 + 256); // per-relation transform
    float*    out1 = (float*)   take((size_t)N * HIDD * 4 + 256);
    float*    out2 = (float*)   take((size_t)N * HIDD * 4 + 256);
    float*    deg  = (float*)   take((size_t)N * NRELD * 4);
    _Float16* W1T  = (_Float16*)take((size_t)NRELD * NBLK * 48 * 64 * 2); // padded B^T blocks
    _Float16* W2T  = (_Float16*)take((size_t)NRELD * NBLK * 48 * 64 * 2);
    _Float16* r1T  = (_Float16*)take((size_t)208 * 224 * 2);              // padded root^T
    _Float16* r2T  = (_Float16*)take((size_t)208 * 224 * 2);
    if (used > ws_size) return; // workspace too small: bail deterministically

    const long nh = (long)N * HIDD;
    const int  TB = 256;

    // --- operand prep: f16 copies, zero-padded transposed B operands ---
    rgcn_cvt_f16<<<cdivl(nh, TB), TB, 0, stream>>>(node_emb, x_h, nh);

    const long wtN = (long)NRELD * NBLK * 48 * 64;
    rgcn_fill_f16<<<cdivl(wtN, TB), TB, 0, stream>>>(W1T, wtN);
    rgcn_fill_f16<<<cdivl(wtN, TB), TB, 0, stream>>>(W2T, wtN);
    rgcn_fill_f16<<<cdivl(208 * 224, TB), TB, 0, stream>>>(r1T, 208 * 224);
    rgcn_fill_f16<<<cdivl(208 * 224, TB), TB, 0, stream>>>(r2T, 208 * 224);

    dim3 wtg((unsigned)cdivl(BSZ * BSZ, TB), 1, NRELD * NBLK);
    rgcn_transpose_cvt<<<wtg, TB, 0, stream>>>(W1, W1T, BSZ, BSZ, 64, BSZ * BSZ, 48 * 64);
    rgcn_transpose_cvt<<<wtg, TB, 0, stream>>>(W2, W2T, BSZ, BSZ, 64, BSZ * BSZ, 48 * 64);
    dim3 rtg((unsigned)cdivl(HIDD * HIDD, TB), 1, 1);
    rgcn_transpose_cvt<<<rtg, TB, 0, stream>>>(root1, r1T, HIDD, HIDD, 224, 0, 0);
    rgcn_transpose_cvt<<<rtg, TB, 0, stream>>>(root2, r2T, HIDD, HIDD, 224, 0, 0);

    // --- per-(dst, relation) degree (shared by both layers) ---
    rgcn_fill_f32<<<cdivl((long)N * NRELD, TB), TB, 0, stream>>>(deg, 0.0f, (long)N * NRELD);
    rgcn_deg_count<<<cdivl(E, TB), TB, 0, stream>>>(dstv, edge_type, deg, E);

    const int mTiles   = N / 16;                 // 100000 % 16 == 0
    const long edgeWGs = cdivl(E, 8);            // 8 waves per 256-thread block
    dim3 tGrid((unsigned)mTiles, 1, NBLK);       // NT=3: full 40-col block per wave
    dim3 rGrid((unsigned)mTiles, 1, 1);          // NT=13: full 200-col strip per wave

    // ---------------- layer 1 ----------------
    rgcn_init_bias<<<cdivl(nh, TB), TB, 0, stream>>>(out1, b1, nh);
    for (int r = 0; r < NRELD; ++r) {
        rgcn_wmma_gemm_f16<3><<<tGrid, 32, 0, stream>>>(
            x_h, HIDD, 0, W1T + (size_t)r * NBLK * 48 * 64, 64,
            Hr, HIDD, 0, BSZ, 64, /*acc=*/0, /*zA=*/BSZ, /*zBT=*/48 * 64, /*zC=*/BSZ);
        rgcn_scatter_rel<<<edgeWGs, TB, 0, stream>>>(srcv, dstv, edge_type, deg, Hr, out1, r, E);
    }
    rgcn_wmma_gemm_f16<13><<<rGrid, 32, 0, stream>>>(
        x_h, HIDD, 0, r1T, 224, out1, HIDD, 0, HIDD, 224, /*acc=*/1, 0, 0, 0);
    rgcn_relu_cvt<<<cdivl(nh, TB), TB, 0, stream>>>(out1, x_h, nh); // z1 -> f16 input

    // ---------------- layer 2 ----------------
    rgcn_init_bias<<<cdivl(nh, TB), TB, 0, stream>>>(out2, b2, nh);
    for (int r = 0; r < NRELD; ++r) {
        rgcn_wmma_gemm_f16<3><<<tGrid, 32, 0, stream>>>(
            x_h, HIDD, 0, W2T + (size_t)r * NBLK * 48 * 64, 64,
            Hr, HIDD, 0, BSZ, 64, 0, BSZ, 48 * 64, BSZ);
        rgcn_scatter_rel<<<edgeWGs, TB, 0, stream>>>(srcv, dstv, edge_type, deg, Hr, out2, r, E);
    }
    rgcn_wmma_gemm_f16<13><<<rGrid, 32, 0, stream>>>(
        x_h, HIDD, 0, r2T, 224, out2, HIDD, 0, HIDD, 224, 1, 0, 0, 0);

    // ---------------- DistMult decode ----------------
    rgcn_decode<<<edgeWGs, TB, 0, stream>>>(srcv, dstv, edge_type, out2, rel_emb,
                                            (float*)d_out, E);
}
